// block_49417893708165
// MI455X (gfx1250) — matrix operand order repaired and checked by reading, not compiled, not run on hardware
//
#include <hip/hip_runtime.h>
#include <math.h>
#include <stdint.h>

// ---------------------------------------------------------------------------
// Types for CDNA5 WMMA (gfx1250, wave32)
// ---------------------------------------------------------------------------
typedef __attribute__((ext_vector_type(16))) __bf16 v16bf;
typedef __attribute__((ext_vector_type(8)))  __bf16 v8bf;
typedef __attribute__((ext_vector_type(8)))  float  v8f;

typedef unsigned short u16;
typedef unsigned int   u32;

__device__ __forceinline__ __bf16 f2bf(float f) {
    union { float f; u32 u; } x; x.f = f;
    u32 r = x.u + 0x7FFFu + ((x.u >> 16) & 1u);   // round-to-nearest-even
    union { u16 u; __bf16 b; } y; y.u = (u16)(r >> 16);
    return y.b;
}

// Load a 16-element bf16 WMMA operand fragment from two 16B chunks
// separated by `stride2` elements (8 for both A and B frags here: 16 contiguous elements).
__device__ __forceinline__ v16bf load_frag(const __bf16* p, int stride2) {
    v8bf lo = *(const v8bf*)(p);
    v8bf hi = *(const v8bf*)(p + stride2);
    v16bf r;
#pragma unroll
    for (int i = 0; i < 8; ++i) { r[i] = lo[i]; r[i + 8] = hi[i]; }
    return r;
}

// ---------------------------------------------------------------------------
// Problem constants
// ---------------------------------------------------------------------------
#define BN   2
#define SEQ  2048
#define DM   1024
#define NH   16
#define DK   64
#define DFF  4096
#define MTOK (BN * SEQ)      // 4096 token rows
#define QT   (SEQ / 16)      // 128 q-tiles per (b,h)

// ---------------------------------------------------------------------------
// Weight transpose + fp32->bf16 convert:  W[K,N] -> WT[N,K]
// ---------------------------------------------------------------------------
__global__ void wtrans_kernel(const float* __restrict__ W, __bf16* __restrict__ WT,
                              int K, int N) {
    int idx = blockIdx.x * 256 + threadIdx.x;
    if (idx < K * N) {
        int k = idx / N;
        int n = idx - k * N;
        WT[(size_t)n * K + k] = f2bf(W[idx]);
    }
}

// ---------------------------------------------------------------------------
// LayerNorm over last dim (D=1024), one block per token row, bf16 out
// ---------------------------------------------------------------------------
__global__ void ln_kernel(const float* __restrict__ x, const float* __restrict__ g,
                          const float* __restrict__ b, __bf16* __restrict__ out, int D) {
    int row = blockIdx.x;
    int tid = threadIdx.x;
    const float* xr = x + (size_t)row * D;
    float s = 0.f, s2 = 0.f;
    for (int i = tid; i < D; i += 256) { float v = xr[i]; s += v; s2 += v * v; }
    __shared__ float rs[256], rq[256];
    rs[tid] = s; rq[tid] = s2;
    __syncthreads();
    for (int off = 128; off > 0; off >>= 1) {
        if (tid < off) { rs[tid] += rs[tid + off]; rq[tid] += rq[tid + off]; }
        __syncthreads();
    }
    float mu  = rs[0] / (float)D;
    float var = rq[0] / (float)D - mu * mu;
    float inv = rsqrtf(var + 1e-5f);
    for (int i = tid; i < D; i += 256)
        out[(size_t)row * D + i] = f2bf((xr[i] - mu) * inv * g[i] + b[i]);
}

// ---------------------------------------------------------------------------
// WMMA GEMM:  C[M,N] = A[M,K](bf16) * BT[N,K](bf16)^T + bias
//   MODE 0: outH = bf16((acc+bias)*scale)                 token-major [M,N]
//   MODE 1: outH = bf16(gelu_exact(acc+bias))             token-major [M,N]
//   MODE 2: outF = res + acc + bias                       fp32 token-major
//   MODE 3: outH = bf16(acc+bias), stored transposed as vT[b,h,d,s]
// N, K, MODE compile-time: tile indexing is shifts, B-row addresses fold into
// one base pointer + immediate offsets, and the k-loop is register
// double-buffered (K % 64 == 0) so next-step fragment loads are in flight
// while the current 4 WMMAs execute.
// ---------------------------------------------------------------------------
template <int N, int K, int MODE>
__global__ void gemm_kernel(const __bf16* __restrict__ A, const __bf16* __restrict__ BT,
                            const float* __restrict__ bias, const float* __restrict__ res,
                            float* __restrict__ outF, __bf16* __restrict__ outH,
                            float scale) {
    constexpr int nTN = N >> 6;
    int wave = (blockIdx.x << 2) + (threadIdx.x >> 5);
    int lane = threadIdx.x & 31;
    int half = lane >> 4;        // 0: lanes 0-15, 1: lanes 16-31
    int l15  = lane & 15;
    int tm   = (wave / nTN) << 4;          // nTN is a power of two -> shift
    int tn   = (wave % nTN) << 6;

    v8f acc[4] = {};
    const __bf16* Arow  = A  + (size_t)(tm + l15) * K + (half << 4);
    const __bf16* Bbase = BT + (size_t)(tn + l15) * K + (half << 4);

    // stage 0 preload
    v16bf a0 = load_frag(Arow, 8);
    v16bf b0[4], b1[4];
#pragma unroll
    for (int t = 0; t < 4; ++t) b0[t] = load_frag(Bbase + (size_t)(t * 16) * K, 8);

    for (int k0 = 0; k0 < K; k0 += 64) {
        // preload stage 1 (k0+32) before consuming stage 0
        v16bf a1 = load_frag(Arow + k0 + 32, 8);
#pragma unroll
        for (int t = 0; t < 4; ++t)
            b1[t] = load_frag(Bbase + (size_t)(t * 16) * K + k0 + 32, 8);

#pragma unroll
        for (int t = 0; t < 4; ++t)
            acc[t] = __builtin_amdgcn_wmma_f32_16x16x32_bf16(
                false, a0, false, b0[t], (short)0, acc[t], false, false);

        int kn = k0 + 64;
        if (kn < K) {
            a0 = load_frag(Arow + kn, 8);
#pragma unroll
            for (int t = 0; t < 4; ++t)
                b0[t] = load_frag(Bbase + (size_t)(t * 16) * K + kn, 8);
        }

#pragma unroll
        for (int t = 0; t < 4; ++t)
            acc[t] = __builtin_amdgcn_wmma_f32_16x16x32_bf16(
                false, a1, false, b1[t], (short)0, acc[t], false, false);
    }

#pragma unroll
    for (int t = 0; t < 4; ++t) {
        int col = tn + (t << 4) + l15;
        float bs = bias[col];
#pragma unroll
        for (int r = 0; r < 8; ++r) {
            int row = tm + r + (half << 3);
            float v = acc[t][r] + bs;
            if (MODE == 0) {
                outH[(size_t)row * N + col] = f2bf(v * scale);
            } else if (MODE == 1) {
                outH[(size_t)row * N + col] =
                    f2bf(0.5f * v * (1.0f + erff(v * 0.70710678118654752f)));
            } else if (MODE == 2) {
                size_t o = (size_t)row * N + col;
                outF[o] = res[o] + v;
            } else {  // MODE 3: V projection stored transposed vT[b,h,d,s]
                int bb = row >> 11;            // row / SEQ
                int ss = row & (SEQ - 1);
                int hh = col >> 6;             // col / DK
                int dd = col & (DK - 1);
                outH[(((size_t)bb * NH + hh) * DK + dd) * SEQ + ss] = f2bf(v);
            }
        }
    }
}

// ---------------------------------------------------------------------------
// Causal flash attention.
//   q,k : bf16 [B,S,H,64]  (q pre-scaled by 1/sqrt(dk))
//   vT  : bf16 [B,H,64,S]  (transposed by the V-projection GEMM epilogue)
// One wave handles one (b, h, 16-row q-tile); streams 32-key chunks with
// online softmax. Per chunk the 64x32 V^T tile is staged into LDS with the
// CDNA5 async-to-LDS DMA (issued before the score WMMAs so it overlaps the
// softmax), then P.V B-fragments are contiguous ds_load_b128 reads.
// Probabilities are re-shaped C-layout -> A-layout through LDS.
// ---------------------------------------------------------------------------
__global__ void attn_kernel(const __bf16* __restrict__ q, const __bf16* __restrict__ k,
                            const __bf16* __restrict__ vT, __bf16* __restrict__ o) {
    int wave = (blockIdx.x << 2) + (threadIdx.x >> 5);
    int lane = threadIdx.x & 31;
    int half = lane >> 4;
    int l15  = lane & 15;

    int b  = wave / (NH * QT);
    int r0 = wave - b * (NH * QT);
    int h  = r0 / QT;
    int qt = r0 - h * QT;

    __shared__ __align__(16) __bf16 pb[4][16 * 32];       // prob tiles, 1KB/wave
    __shared__ __align__(16) __bf16 vs[4][64 * 32];       // V^T tiles,  4KB/wave
    __bf16* pl    = pb[threadIdx.x >> 5];
    __bf16* vtile = vs[threadIdx.x >> 5];
    unsigned vtile_lds = (unsigned)(uintptr_t)vtile;      // 32-bit LDS offset

    // Q A-fragments (two k-steps over DK=64), held in registers
    v16bf aq[2];
    {
        const __bf16* qp = q + ((size_t)(b * SEQ + qt * 16 + l15)) * DM + h * DK + (half << 4);
#pragma unroll
        for (int c = 0; c < 2; ++c) aq[c] = load_frag(qp + c * 32, 8);
    }

    float mrow[8], lrow[8];
#pragma unroll
    for (int r = 0; r < 8; ++r) { mrow[r] = -1e30f; lrow[r] = 0.f; }
    v8f accO[4] = {};

    int nch = (qt * 16 + 16 + 31) >> 5;   // causal: chunks of 32 keys

    const __bf16* vbase = vT + ((size_t)(b * NH + h) * DK) * SEQ;  // row d=0

    for (int ch = 0; ch < nch; ++ch) {
        // ---- async DMA of the 64x32 V^T tile (64 d-rows x 64B) into LDS ---
        {
            // prior-chunk DS reads of vtile must complete before overwrite
            asm volatile("s_wait_dscnt 0x0" ::: "memory");
#pragma unroll
            for (int i = 0; i < 8; ++i) {
                int chunk = i * 32 + lane;        // 256 x 16B chunks
                int d     = chunk >> 2;           // 0..63
                int c16   = chunk & 3;            // 16B chunk within row
                unsigned long long ga = (unsigned long long)(uintptr_t)
                    (vbase + (size_t)d * SEQ + ch * 32 + c16 * 8);
                unsigned dst = vtile_lds + (unsigned)(d * 64 + c16 * 16);
                asm volatile("global_load_async_to_lds_b128 %0, %1, off"
                             :: "v"(dst), "v"(ga) : "memory");
            }
        }

        // ---- scores: Q(16x64) x K_chunk^T(64x32) -> two 16x16 tiles ----
        v8f sc[2] = {};
#pragma unroll
        for (int kt = 0; kt < 2; ++kt) {
            int ncol = ch * 32 + kt * 16 + l15;                   // key token
            const __bf16* kp = k + ((size_t)(b * SEQ + ncol)) * DM + h * DK + (half << 4);
#pragma unroll
            for (int c = 0; c < 2; ++c) {
                v16bf bb = load_frag(kp + c * 32, 8);
                sc[kt] = __builtin_amdgcn_wmma_f32_16x16x32_bf16(
                    false, aq[c], false, bb, (short)0, sc[kt], false, false);
            }
        }

        // ---- causal mask + online softmax (rows in VGPR idx, cols in lanes)
#pragma unroll
        for (int r = 0; r < 8; ++r) {
            int qrow = qt * 16 + r + (half << 3);
            float mx = -1e30f;
#pragma unroll
            for (int kt = 0; kt < 2; ++kt) {
                int ncol = ch * 32 + kt * 16 + l15;
                if (ncol > qrow) sc[kt][r] = -1e30f;
                mx = fmaxf(mx, sc[kt][r]);
            }
            // reduce within each 16-lane half (xor of bits 0-3 stays in-half)
            for (int off = 1; off < 16; off <<= 1)
                mx = fmaxf(mx, __shfl_xor(mx, off, 32));
            float nm   = fmaxf(mrow[r], mx);
            float corr = __expf(mrow[r] - nm);
            mrow[r] = nm;
            float rsum = 0.f;
#pragma unroll
            for (int kt = 0; kt < 2; ++kt) {
                float p = __expf(sc[kt][r] - nm);
                sc[kt][r] = p;
                rsum += p;
            }
            for (int off = 1; off < 16; off <<= 1)
                rsum += __shfl_xor(rsum, off, 32);
            lrow[r] = lrow[r] * corr + rsum;
#pragma unroll
            for (int t = 0; t < 4; ++t) accO[t][r] *= corr;
        }

        // ---- probs C-layout -> A-fragment layout through LDS ----
#pragma unroll
        for (int kt = 0; kt < 2; ++kt)
#pragma unroll
            for (int r = 0; r < 8; ++r)
                pl[(r + (half << 3)) * 32 + kt * 16 + l15] = f2bf(sc[kt][r]);
        __builtin_amdgcn_wave_barrier();

        v16bf ap = load_frag(pl + l15 * 32 + (half << 4), 8);

        // ---- drain async V DMA, then P(16x32) x V_chunk(32x64) ----
        // B-frag element e: key=(half*16+e), d = vt*16+l15 -> contiguous in
        // the LDS V^T tile row: 2 x ds_load_b128 per fragment.
        asm volatile("s_wait_asynccnt 0x0" ::: "memory");
        __builtin_amdgcn_wave_barrier();
#pragma unroll
        for (int vt = 0; vt < 4; ++vt) {
            const __bf16* vp = vtile + (size_t)((vt << 4) + l15) * 32 + (half << 4);
            v16bf bv = load_frag(vp, 8);
            accO[vt] = __builtin_amdgcn_wmma_f32_16x16x32_bf16(
                false, ap, false, bv, (short)0, accO[vt], false, false);
        }
    }

    // ---- normalize and store bf16 [B,S,H,64] ----
#pragma unroll
    for (int vt = 0; vt < 4; ++vt) {
        int d = (vt << 4) + l15;
#pragma unroll
        for (int r = 0; r < 8; ++r) {
            int qrow = qt * 16 + r + (half << 3);
            o[((size_t)(b * SEQ + qrow)) * DM + h * DK + d] = f2bf(accO[vt][r] / lrow[r]);
        }
    }
}

// ---------------------------------------------------------------------------
// Host launch
// ---------------------------------------------------------------------------
extern "C" void kernel_launch(void* const* d_in, const int* in_sizes, int n_in,
                              void* d_out, int out_size, void* d_ws, size_t ws_size,
                              hipStream_t stream) {
    const float* x    = (const float*)d_in[0];
    const float* ln1g = (const float*)d_in[1];
    const float* ln1b = (const float*)d_in[2];
    const float* Wq   = (const float*)d_in[3];
    const float* bq   = (const float*)d_in[4];
    const float* Wk   = (const float*)d_in[5];
    const float* bk   = (const float*)d_in[6];
    const float* Wv   = (const float*)d_in[7];
    const float* bv_  = (const float*)d_in[8];
    const float* Wo   = (const float*)d_in[9];
    const float* bo   = (const float*)d_in[10];
    const float* ln2g = (const float*)d_in[11];
    const float* ln2b = (const float*)d_in[12];
    const float* W1   = (const float*)d_in[13];
    const float* b1   = (const float*)d_in[14];
    const float* W2   = (const float*)d_in[15];
    const float* b2   = (const float*)d_in[16];
    float* out = (float*)d_out;

    size_t off = 0;
    auto alloc = [&](size_t bytes) -> void* {
        void* p = (char*)d_ws + off;
        off += (bytes + 255) & ~(size_t)255;
        return p;
    };
    __bf16* WqT = (__bf16*)alloc((size_t)DM * DM * 2);
    __bf16* WkT = (__bf16*)alloc((size_t)DM * DM * 2);
    __bf16* WvT = (__bf16*)alloc((size_t)DM * DM * 2);
    __bf16* WoT = (__bf16*)alloc((size_t)DM * DM * 2);
    __bf16* W1T = (__bf16*)alloc((size_t)DM * DFF * 2);
    __bf16* W2T = (__bf16*)alloc((size_t)DFF * DM * 2);
    __bf16* h1  = (__bf16*)alloc((size_t)MTOK * DM * 2);
    __bf16* qb  = (__bf16*)alloc((size_t)MTOK * DM * 2);
    __bf16* kb  = (__bf16*)alloc((size_t)MTOK * DM * 2);
    __bf16* vtb = (__bf16*)alloc((size_t)MTOK * DM * 2);   // V^T [B,H,64,S]
    __bf16* ao  = (__bf16*)alloc((size_t)MTOK * DM * 2);
    float*  x2  = (float*) alloc((size_t)MTOK * DM * 4);
    __bf16* h2  = (__bf16*)alloc((size_t)MTOK * DM * 2);
    __bf16* hf  = (__bf16*)alloc((size_t)MTOK * DFF * 2);

    // 1) weights -> bf16 transposed [N,K]
    int nw = DM * DM;
    wtrans_kernel<<<(nw + 255) / 256, 256, 0, stream>>>(Wq, WqT, DM, DM);
    wtrans_kernel<<<(nw + 255) / 256, 256, 0, stream>>>(Wk, WkT, DM, DM);
    wtrans_kernel<<<(nw + 255) / 256, 256, 0, stream>>>(Wv, WvT, DM, DM);
    wtrans_kernel<<<(nw + 255) / 256, 256, 0, stream>>>(Wo, WoT, DM, DM);
    int nw1 = DM * DFF;
    wtrans_kernel<<<(nw1 + 255) / 256, 256, 0, stream>>>(W1, W1T, DM, DFF);
    wtrans_kernel<<<(nw1 + 255) / 256, 256, 0, stream>>>(W2, W2T, DFF, DM);

    // 2) LN1
    ln_kernel<<<MTOK, 256, 0, stream>>>(x, ln1g, ln1b, h1, DM);

    // 3) QKV projections (q pre-scaled by 1/sqrt(dk); V stored transposed)
    int gD = (MTOK / 16) * (DM / 64) / 4;      // 1024 blocks
    gemm_kernel<DM, DM, 0><<<gD, 128, 0, stream>>>(h1, WqT, bq, nullptr, nullptr, qb, 0.125f);
    gemm_kernel<DM, DM, 0><<<gD, 128, 0, stream>>>(h1, WkT, bk, nullptr, nullptr, kb, 1.0f);
    gemm_kernel<DM, DM, 3><<<gD, 128, 0, stream>>>(h1, WvT, bv_, nullptr, nullptr, vtb, 1.0f);

    // 4) causal attention
    attn_kernel<<<(BN * NH * QT) / 4, 128, 0, stream>>>(qb, kb, vtb, ao);

    // 5) output projection + residual (fp32)
    gemm_kernel<DM, DM, 2><<<gD, 128, 0, stream>>>(ao, WoT, bo, x, x2, nullptr, 1.0f);

    // 6) LN2 + FFN
    ln_kernel<<<MTOK, 256, 0, stream>>>(x2, ln2g, ln2b, h2, DM);
    int gF1 = (MTOK / 16) * (DFF / 64) / 4;    // 4096 blocks
    gemm_kernel<DFF, DM, 1><<<gF1, 128, 0, stream>>>(h2, W1T, b1, nullptr, nullptr, hf, 1.0f);
    gemm_kernel<DM, DFF, 2><<<gD, 128, 0, stream>>>(hf, W2T, b2, x2, out, nullptr, 1.0f);

    (void)in_sizes; (void)n_in; (void)out_size; (void)ws_size;
}
